// GeometricSpatialAttention_49263274885650
// MI455X (gfx1250) — matrix operand order repaired and checked by reading, not compiled, not run on hardware
//
#include <hip/hip_runtime.h>
#include <hip/hip_bf16.h>
#include <math.h>

typedef __attribute__((ext_vector_type(16))) _Float16 v16h;
typedef __attribute__((ext_vector_type(8)))  float    v8f;
typedef __attribute__((ext_vector_type(2)))  _Float16 h2;

#define N_TOK   2000
#define E_DIM   512
#define H_HEADS 8
#define DH      64
#define K_NN    32
#define CLAMP_V 10000.0f
#define LN_EPS  1e-5f

// ---------------------------------------------------------------------------
// Kernel 1: centroids + pos-encoder (Linear(3,128)+GELU+LayerNorm) fused with
// prompt-add and f16 conversion of the 2000x512 activation matrix.
// One 128-thread block per row.
// ---------------------------------------------------------------------------
__global__ __launch_bounds__(128) void prep_kernel(
    const float* __restrict__ x, const float* __restrict__ coords9,
    const float* __restrict__ prompt, const float* __restrict__ pos_w1,
    const float* __restrict__ pos_b1, const float* __restrict__ ln_g,
    const float* __restrict__ ln_b, float* __restrict__ centroids,
    _Float16* __restrict__ Xh) {
  const int n = blockIdx.x;
  const int t = threadIdx.x;  // 0..127
  __shared__ float c[3];
  __shared__ float red[128];

  if (t < 3) {
    float v = (coords9[n * 9 + t] + coords9[n * 9 + 3 + t] +
               coords9[n * 9 + 6 + t]) * (1.0f / 3.0f);
    c[t] = v;
    centroids[n * 3 + t] = v;
  }
  __syncthreads();

  float hpre = c[0] * pos_w1[t * 3 + 0] + c[1] * pos_w1[t * 3 + 1] +
               c[2] * pos_w1[t * 3 + 2] + pos_b1[t];
  // exact GELU: 0.5*x*(1+erf(x/sqrt(2)))
  float g = 0.5f * hpre * (1.0f + erff(hpre * 0.70710678118654752f));

  red[t] = g;
  __syncthreads();
  for (int s = 64; s > 0; s >>= 1) { if (t < s) red[t] += red[t + s]; __syncthreads(); }
  float mu = red[0] * (1.0f / 128.0f);
  __syncthreads();
  float d = g - mu;
  red[t] = d * d;
  __syncthreads();
  for (int s = 64; s > 0; s >>= 1) { if (t < s) red[t] += red[t + s]; __syncthreads(); }
  float var = red[0] * (1.0f / 128.0f);
  float y = d * rsqrtf(var + LN_EPS) * ln_g[t] + ln_b[t];

  Xh[(size_t)n * E_DIM + 384 + t] = (_Float16)y;
  for (int i = t; i < 384; i += 128)
    Xh[(size_t)n * E_DIM + i] = (_Float16)(x[(size_t)n * E_DIM + i] + prompt[i]);
}

// ---------------------------------------------------------------------------
// Kernel 2: per-row top-K (K=32) nearest centroids. One 256-thread block per
// row; squared distances cached in LDS, 32 argmin passes with LDS tree
// reduction (deterministic tie-break to lowest index).
// ---------------------------------------------------------------------------
__global__ __launch_bounds__(256) void topk_kernel(
    const float* __restrict__ centroids, int* __restrict__ topk) {
  const int i = blockIdx.x;
  const int t = threadIdx.x;  // 0..255
  __shared__ float dist[2048];
  __shared__ float rv[256];
  __shared__ int   ri[256];

  const float cx = centroids[i * 3 + 0];
  const float cy = centroids[i * 3 + 1];
  const float cz = centroids[i * 3 + 2];

  for (int j = t; j < 2048; j += 256) {
    float d2 = 3.0e38f;
    if (j < N_TOK) {
      float dx = cx - centroids[j * 3 + 0];
      float dy = cy - centroids[j * 3 + 1];
      float dz = cz - centroids[j * 3 + 2];
      d2 = dx * dx + dy * dy + dz * dz;  // sqrt is monotone -> skip it
    }
    dist[j] = d2;
  }
  __syncthreads();

  for (int k = 0; k < K_NN; ++k) {
    float best = 3.0e38f;
    int bidx = 0x7fffffff;
    for (int j = t; j < 2048; j += 256) {
      float v = dist[j];
      if (v < best || (v == best && j < bidx)) { best = v; bidx = j; }
    }
    rv[t] = best; ri[t] = bidx;
    __syncthreads();
    for (int s = 128; s > 0; s >>= 1) {
      if (t < s) {
        float ov = rv[t + s]; int oi = ri[t + s];
        if (ov < rv[t] || (ov == rv[t] && oi < ri[t])) { rv[t] = ov; ri[t] = oi; }
      }
      __syncthreads();
    }
    if (t == 0) {
      topk[i * K_NN + k] = ri[0];
      dist[ri[0]] = 3.0e38f;
    }
    __syncthreads();
  }
}

// ---------------------------------------------------------------------------
// Kernel 3: fp32 -> f16 weight conversion (one-time; weights stay L2-hot).
// ---------------------------------------------------------------------------
__global__ void f32_to_f16_kernel(const float* __restrict__ src,
                                  _Float16* __restrict__ dst, int n) {
  int i = blockIdx.x * blockDim.x + threadIdx.x;
  if (i < n) dst[i] = (_Float16)src[i];
}

// ---------------------------------------------------------------------------
// Kernel 4: WMMA GEMM  out[m, o] = sum_k Ah[m,k] * Wh[o,k] + bias[o]
// Ah: Mx512 f16 row-major, Wh: 512x512 f16 row-major (out, in) -- both
// contract along the contiguous axis, so WMMA fragment pair-loads are
// contiguous (mergeable into b128 loads).
// Grid (125 M-tiles, 2 col-strips of 256), 256 threads = 8 waves. Each wave
// register-blocks TWO 16x16 output tiles (16x32), reusing one A fragment per
// pair of v_wmma_f32_16x16x32_f16 -> 6 b128 loads per 2 WMMAs.
// ---------------------------------------------------------------------------
__global__ __launch_bounds__(256) void gemm512_wmma_kernel(
    const _Float16* __restrict__ Ah, const _Float16* __restrict__ Wh,
    const float* __restrict__ bias, float* __restrict__ out, int final_clamp) {
  const int lane = threadIdx.x & 31;
  const int wave = threadIdx.x >> 5;
  const int half = lane >> 4;
  const int l15  = lane & 15;
  const int row0 = blockIdx.x * 16;
  const int col0 = blockIdx.y * 256 + wave * 32 + l15;  // lane's col, tile 0
  const int col1 = col0 + 16;                           // lane's col, tile 1
  const int m    = row0 + l15;                          // lane's A row

  const _Float16* aRow  = Ah + (size_t)m * E_DIM;
  const _Float16* bRow0 = Wh + (size_t)col0 * E_DIM;
  const _Float16* bRow1 = Wh + (size_t)col1 * E_DIM;

  v8f acc0 = {};
  v8f acc1 = {};
#pragma unroll 4
  for (int k0 = 0; k0 < E_DIM; k0 += 32) {
    v16h A, B0, B1;
#pragma unroll
    for (int v = 0; v < 8; ++v) {
      // A 16x32 f16 layout (ISA 7.12.2): lanes 0-15 K={2v|16+2(v-4)},
      // lanes 16-31 same +8.
      const int kbA = (v < 4 ? 2 * v : 16 + 2 * (v - 4)) + 8 * half;
      h2 pa = *(const h2*)(aRow + k0 + kbA);
      A[2 * v + 0] = pa[0];
      A[2 * v + 1] = pa[1];
      // B 32x16 f16: same K order as A (lanes 0-15 K={2v|16+2(v-4)}, lanes 16-31 +8).
      const int kbB = (v < 4 ? 2 * v : 16 + 2 * (v - 4)) + 8 * half;
      h2 pb0 = *(const h2*)(bRow0 + k0 + kbB);
      B0[2 * v + 0] = pb0[0];
      B0[2 * v + 1] = pb0[1];
      h2 pb1 = *(const h2*)(bRow1 + k0 + kbB);
      B1[2 * v + 0] = pb1[0];
      B1[2 * v + 1] = pb1[1];
    }
    if (k0 + 32 < E_DIM) {
      // locality=3 -> WGP-scope prefetch (pull into all cache levels)
      __builtin_prefetch(aRow + k0 + 32, 0, 3);
      __builtin_prefetch(bRow0 + k0 + 32, 0, 3);
      __builtin_prefetch(bRow1 + k0 + 32, 0, 3);
    }
    acc0 = __builtin_amdgcn_wmma_f32_16x16x32_f16(
        false, A, false, B0, (short)0, acc0, false, false);
    acc1 = __builtin_amdgcn_wmma_f32_16x16x32_f16(
        false, A, false, B1, (short)0, acc1, false, false);
  }

  const float b0 = bias[col0];
  const float b1 = bias[col1];
#pragma unroll
  for (int r = 0; r < 8; ++r) {
    const int row = row0 + r + 8 * half;  // D layout: VGPR r -> M = r + 8*half
    float v0 = acc0[r] + b0;
    float v1 = acc1[r] + b1;
    if (final_clamp) {
      if (v0 != v0) v0 = 0.0f;                       // nan_to_num
      if (v1 != v1) v1 = 0.0f;
      v0 = fminf(fmaxf(v0, -CLAMP_V), CLAMP_V);      // +/-inf clamp too
      v1 = fminf(fmaxf(v1, -CLAMP_V), CLAMP_V);
    }
    out[(size_t)row * E_DIM + col0] = v0;
    out[(size_t)row * E_DIM + col1] = v1;
  }
}

// ---------------------------------------------------------------------------
// Kernel 5: sparse top-K attention. One 64-thread block per (row, head).
// Wave 0 lanes each own one of the 32 neighbors: dot(q,k)/8, softmax via
// wave32 shuffle reductions; then all 64 threads (one per head dim) gather
// the weighted V sum and emit f16 context.
// ---------------------------------------------------------------------------
__global__ __launch_bounds__(64) void attn_kernel(
    const float* __restrict__ q, const float* __restrict__ k,
    const float* __restrict__ v, const int* __restrict__ topk,
    _Float16* __restrict__ ctx_h) {
  const int i = blockIdx.x;
  const int h = blockIdx.y;
  const int t = threadIdx.x;  // 0..63
  __shared__ float sA[K_NN];
  __shared__ int   sI[K_NN];

  if (t < K_NN) {
    const int nb = topk[i * K_NN + t];
    const float* qp = q + (size_t)i * E_DIM + h * DH;
    const float* kp = k + (size_t)nb * E_DIM + h * DH;
    float dot = 0.0f;
#pragma unroll 8
    for (int d = 0; d < DH; ++d) dot += qp[d] * kp[d];
    float s = dot * 0.125f;  // 1/sqrt(64)

    float mx = s;
    for (int off = 16; off > 0; off >>= 1)
      mx = fmaxf(mx, __shfl_xor(mx, off, 32));
    float p = __expf(s - mx);
    float sum = p;
    for (int off = 16; off > 0; off >>= 1)
      sum += __shfl_xor(sum, off, 32);

    sA[t] = p / sum;
    sI[t] = nb;
  }
  __syncthreads();

  float acc = 0.0f;
#pragma unroll 4
  for (int j = 0; j < K_NN; ++j)
    acc += sA[j] * v[(size_t)sI[j] * E_DIM + h * DH + t];
  ctx_h[(size_t)i * E_DIM + h * DH + t] = (_Float16)acc;
}

// ---------------------------------------------------------------------------
extern "C" void kernel_launch(void* const* d_in, const int* in_sizes, int n_in,
                              void* d_out, int out_size, void* d_ws,
                              size_t ws_size, hipStream_t stream) {
  (void)in_sizes; (void)n_in; (void)out_size; (void)ws_size;
  const float* x       = (const float*)d_in[0];
  const float* coords9 = (const float*)d_in[1];
  const float* prompt  = (const float*)d_in[2];
  const float* pos_w1  = (const float*)d_in[3];
  const float* pos_b1  = (const float*)d_in[4];
  const float* ln_g    = (const float*)d_in[5];
  const float* ln_b    = (const float*)d_in[6];
  const float* wq      = (const float*)d_in[7];
  const float* wk      = (const float*)d_in[8];
  const float* wv      = (const float*)d_in[9];
  const float* bq      = (const float*)d_in[10];
  const float* bk      = (const float*)d_in[11];
  const float* bv      = (const float*)d_in[12];
  const float* wo      = (const float*)d_in[13];
  const float* bo      = (const float*)d_in[14];
  float* out = (float*)d_out;

  char* ws = (char*)d_ws;
  size_t off = 0;
  auto alloc = [&](size_t bytes) -> char* {
    char* p = ws + off;
    off += bytes;
    off = (off + 255) & ~(size_t)255;
    return p;
  };

  float*     centroids = (float*)alloc((size_t)N_TOK * 3 * sizeof(float));
  _Float16*  Xh        = (_Float16*)alloc((size_t)N_TOK * E_DIM * sizeof(_Float16));
  _Float16*  wq_h      = (_Float16*)alloc((size_t)E_DIM * E_DIM * sizeof(_Float16));
  _Float16*  wk_h      = (_Float16*)alloc((size_t)E_DIM * E_DIM * sizeof(_Float16));
  _Float16*  wv_h      = (_Float16*)alloc((size_t)E_DIM * E_DIM * sizeof(_Float16));
  _Float16*  wo_h      = (_Float16*)alloc((size_t)E_DIM * E_DIM * sizeof(_Float16));
  int*       topk      = (int*)alloc((size_t)N_TOK * K_NN * sizeof(int));
  float*     qf        = (float*)alloc((size_t)N_TOK * E_DIM * sizeof(float));
  float*     kf        = (float*)alloc((size_t)N_TOK * E_DIM * sizeof(float));
  float*     vf        = (float*)alloc((size_t)N_TOK * E_DIM * sizeof(float));
  _Float16*  ctx_h     = (_Float16*)alloc((size_t)N_TOK * E_DIM * sizeof(_Float16));

  prep_kernel<<<N_TOK, 128, 0, stream>>>(x, coords9, prompt, pos_w1, pos_b1,
                                         ln_g, ln_b, centroids, Xh);
  topk_kernel<<<N_TOK, 256, 0, stream>>>(centroids, topk);

  const int wn = E_DIM * E_DIM;
  f32_to_f16_kernel<<<(wn + 255) / 256, 256, 0, stream>>>(wq, wq_h, wn);
  f32_to_f16_kernel<<<(wn + 255) / 256, 256, 0, stream>>>(wk, wk_h, wn);
  f32_to_f16_kernel<<<(wn + 255) / 256, 256, 0, stream>>>(wv, wv_h, wn);
  f32_to_f16_kernel<<<(wn + 255) / 256, 256, 0, stream>>>(wo, wo_h, wn);

  dim3 gg(N_TOK / 16, E_DIM / 256);  // 125 x 2
  gemm512_wmma_kernel<<<gg, 256, 0, stream>>>(Xh, wq_h, bq, qf, 0);
  gemm512_wmma_kernel<<<gg, 256, 0, stream>>>(Xh, wk_h, bk, kf, 0);
  gemm512_wmma_kernel<<<gg, 256, 0, stream>>>(Xh, wv_h, bv, vf, 0);

  dim3 ga(N_TOK, H_HEADS);
  attn_kernel<<<ga, 64, 0, stream>>>(qf, kf, vf, topk, ctx_h);

  gemm512_wmma_kernel<<<gg, 256, 0, stream>>>(ctx_h, wo_h, bo, out, 1);
}
